// SelfAttention_48481590837426
// MI455X (gfx1250) — matrix operand order repaired and checked by reading, not compiled, not run on hardware
//
#include <hip/hip_runtime.h>

// ---------------------------------------------------------------------------
// CDNA5 (gfx1250) self-attention: bf16 WMMA (16x16x32) with f32 accumulation.
// Round 2: software-pipelined inner loops (double-buffered fragments) so
// VMEM loads overlap the matrix pipe instead of s_wait_loadcnt 0x0 per WMMA.
// ---------------------------------------------------------------------------

typedef __attribute__((ext_vector_type(16))) __bf16 v16bf;
typedef __attribute__((ext_vector_type(8)))  __bf16 v8bf;
typedef __attribute__((ext_vector_type(4)))  __bf16 v4bf;
typedef __attribute__((ext_vector_type(8)))  float  v8f;
typedef __attribute__((ext_vector_type(4)))  float  v4f;

#define DEV static __device__ __forceinline__

DEV v8f vzero8() {
  v8f z;
  for (int i = 0; i < 8; ++i) z[i] = 0.f;
  return z;
}

// Build a 16-element bf16 fragment from two contiguous 16-byte chunks.
DEV v16bf ld_frag(const __bf16* p0, const __bf16* p1) {
  v8bf lo = *(const v8bf*)p0;
  v8bf hi = *(const v8bf*)p1;
  return __builtin_shufflevector(lo, hi, 0, 1, 2, 3, 4, 5, 6, 7,
                                 8, 9, 10, 11, 12, 13, 14, 15);
}

DEV v8f wmma_bf16(v16bf a, v16bf b, v8f c) {
  // (neg_a, A, neg_b, B, c_mod, C, reuse_a, reuse_b)
  return __builtin_amdgcn_wmma_f32_16x16x32_bf16(false, a, false, b,
                                                 (short)0, c, false, false);
}

// ---------------------------------------------------------------------------
// Kernel 1: f32 -> bf16 conversion of x ([16384,1024], 16.7M elements)
// ---------------------------------------------------------------------------
__global__ __launch_bounds__(256) void conv_x_kernel(const float* __restrict__ x,
                                                     __bf16* __restrict__ xb) {
  const int i = (blockIdx.x * 256 + threadIdx.x) * 4;
  v4f f = *(const v4f*)(x + i);
  v4bf o;
#pragma unroll
  for (int r = 0; r < 4; ++r) o[r] = (__bf16)f[r];
  *(v4bf*)(xb + i) = o;
}

// ---------------------------------------------------------------------------
// Kernel 2: W [K=1024][N=1024] f32  ->  Wt [N][K] bf16 (LDS-tiled transpose)
// ---------------------------------------------------------------------------
__global__ __launch_bounds__(256) void conv_wt_kernel(const float* __restrict__ W,
                                                      __bf16* __restrict__ Wt) {
  __shared__ float tile[32][33];
  const int n0 = blockIdx.x * 32, k0 = blockIdx.y * 32;
  const int tx = threadIdx.x, ty = threadIdx.y;  // 32 x 8
#pragma unroll
  for (int j = 0; j < 4; ++j)
    tile[ty + 8 * j][tx] = W[(size_t)(k0 + ty + 8 * j) * 1024 + n0 + tx];
  __syncthreads();
#pragma unroll
  for (int j = 0; j < 4; ++j)
    Wt[(size_t)(n0 + ty + 8 * j) * 1024 + k0 + tx] = (__bf16)tile[tx][ty + 8 * j];
}

// ---------------------------------------------------------------------------
// Kernel 3: q/k projection.  D = Wt-tile(16h x K) x x^T-tile(K x 16s) + bias.
// C-fragment lane n holds q[s0+n][h0 + rb .. rb+7]  -> contiguous bf16 store.
// scale = 1/sqrt(H) for q, 1.0 for k (folds attention scale into q).
// Inner loop double-buffered: next-K fragments load while current K WMMAs run.
// ---------------------------------------------------------------------------
__global__ __launch_bounds__(256) void qk_proj_kernel(
    const __bf16* __restrict__ xb, const __bf16* __restrict__ Wt,
    const float* __restrict__ bias, __bf16* __restrict__ outq, float scale) {
  const int gw   = blockIdx.x * 8 + (threadIdx.x >> 5);
  const int lane = threadIdx.x & 31;
  const int h0 = (gw & 63) << 4;       // 64 h-tiles
  const int s0 = (gw >> 6) << 6;       // 256 groups of 64 rows
  const int m_ln = lane & 15;
  const int kb8 = (lane >> 4) << 3, kb16 = (lane >> 4) << 4, rb = kb8;

  v8f acc[4];
#pragma unroll
  for (int i = 0; i < 4; ++i) acc[i] = vzero8();

  const __bf16* wrow = Wt + (size_t)(h0 + m_ln) * 1024;
  auto ldA = [&](int d) -> v16bf {
    return ld_frag(wrow + d + kb16, wrow + d + kb16 + 8);
  };
  auto ldB = [&](int i, int d) -> v16bf {
    const __bf16* xr = xb + (size_t)(s0 + i * 16 + m_ln) * 1024 + d + kb16;
    return ld_frag(xr, xr + 8);
  };

  v16bf a = ldA(0);
  v16bf b0 = ldB(0, 0), b1 = ldB(1, 0), b2 = ldB(2, 0), b3 = ldB(3, 0);
  for (int d0 = 0; d0 < 1024; d0 += 32) {
    const int dn = (d0 + 32) & 1023;   // wrap keeps prefetch in-bounds
    v16bf a_n = ldA(dn);
    v16bf b0n = ldB(0, dn), b1n = ldB(1, dn), b2n = ldB(2, dn), b3n = ldB(3, dn);
    acc[0] = wmma_bf16(a, b0, acc[0]);
    acc[1] = wmma_bf16(a, b1, acc[1]);
    acc[2] = wmma_bf16(a, b2, acc[2]);
    acc[3] = wmma_bf16(a, b3, acc[3]);
    a = a_n; b0 = b0n; b1 = b1n; b2 = b2n; b3 = b3n;
  }

  v4f bia0 = *(const v4f*)(bias + h0 + rb);
  v4f bia1 = *(const v4f*)(bias + h0 + rb + 4);
#pragma unroll
  for (int i = 0; i < 4; ++i) {
    const int s = s0 + i * 16 + m_ln;
    v8bf pk;
#pragma unroll
    for (int r = 0; r < 8; ++r) {
      const float bb = (r < 4) ? bia0[r] : bia1[r - 4];
      pk[r] = (__bf16)((acc[i][r] + bb) * scale);
    }
    *(v8bf*)(outq + (size_t)s * 1024 + h0 + rb) = pk;
  }
}

// ---------------------------------------------------------------------------
// Kernel 4: v projection, stored transposed: vT[b][h][s] bf16 (pipelined).
// ---------------------------------------------------------------------------
__global__ __launch_bounds__(256) void v_proj_kernel(
    const __bf16* __restrict__ xb, const __bf16* __restrict__ Wt,
    const float* __restrict__ bias, __bf16* __restrict__ vT) {
  const int gw   = blockIdx.x * 8 + (threadIdx.x >> 5);
  const int lane = threadIdx.x & 31;
  const int m0 = (gw >> 4) << 4;       // 1024 s-tiles (global row in x)
  const int h0 = (gw & 15) << 6;       // 16 groups of 64 h
  const int m_ln = lane & 15;
  const int kb8 = (lane >> 4) << 3, kb16 = (lane >> 4) << 4, rb = kb8;

  v8f acc[4];
#pragma unroll
  for (int i = 0; i < 4; ++i) acc[i] = vzero8();

  const __bf16* xrow = xb + (size_t)(m0 + m_ln) * 1024;
  auto ldA = [&](int d) -> v16bf {
    return ld_frag(xrow + d + kb16, xrow + d + kb16 + 8);
  };
  auto ldB = [&](int i, int d) -> v16bf {
    const __bf16* wr = Wt + (size_t)(h0 + i * 16 + m_ln) * 1024 + d + kb16;
    return ld_frag(wr, wr + 8);
  };

  v16bf a = ldA(0);
  v16bf b0 = ldB(0, 0), b1 = ldB(1, 0), b2 = ldB(2, 0), b3 = ldB(3, 0);
  for (int d0 = 0; d0 < 1024; d0 += 32) {
    const int dn = (d0 + 32) & 1023;
    v16bf a_n = ldA(dn);
    v16bf b0n = ldB(0, dn), b1n = ldB(1, dn), b2n = ldB(2, dn), b3n = ldB(3, dn);
    acc[0] = wmma_bf16(a, b0, acc[0]);
    acc[1] = wmma_bf16(a, b1, acc[1]);
    acc[2] = wmma_bf16(a, b2, acc[2]);
    acc[3] = wmma_bf16(a, b3, acc[3]);
    a = a_n; b0 = b0n; b1 = b1n; b2 = b2n; b3 = b3n;
  }

  const int b = m0 >> 11;          // /2048
  const int sl = m0 & 2047;
#pragma unroll
  for (int i = 0; i < 4; ++i) {
    const int h = h0 + i * 16 + m_ln;
    const float bv = bias[h];
    v8bf pk;
#pragma unroll
    for (int r = 0; r < 8; ++r) pk[r] = (__bf16)(acc[i][r] + bv);
    *(v8bf*)(vT + ((size_t)(b * 1024 + h)) * 2048 + sl + rb) = pk;
  }
}

// ---------------------------------------------------------------------------
// Kernel 5: fused attention. One workgroup (8 waves) per (batch, 16 q-rows).
//   phase 1: scores = q(LDS) x k^T  -> bf16 in LDS probs[16][2048+pad]
//            (two j-tiles per A-fragment, pipelined loads)
//   phase 2: row softmax (f32 max/exp/sum), exp stored in place, 1/sum kept
//   phase 3: out = (probs x v) * rowinv, via vT (pipelined A + B ping-pong)
// LDS: 33,024 + 65,792 + 1,152 = 99,968 B per workgroup.
// ---------------------------------------------------------------------------
__global__ __launch_bounds__(256) void attn_fused_kernel(
    const __bf16* __restrict__ qg, const __bf16* __restrict__ kg,
    const __bf16* __restrict__ vTg, float* __restrict__ outg) {
  struct __align__(16) Smem {
    __bf16 qtile[16][1032];   // padded: stride 2064 B (conflict-free b128)
    __bf16 probs[16][2056];   // padded: stride 4112 B (conflict-free b128)
    float  red[16][16];
    float  rowmax[16];
    float  rowinv[16];
  };
  __shared__ Smem sm;

  const int tid  = threadIdx.x;
  const int lane = tid & 31;
  const int w    = tid >> 5;
  const int b    = blockIdx.x >> 7;
  const int s0   = (blockIdx.x & 127) << 4;

  const int m_ln = lane & 15;
  const int kb8  = (lane >> 4) << 3;
  const int kb16 = (lane >> 4) << 4;
  const int rb   = kb8;

  // ---- stage q tile (already scaled by 1/sqrt(H)) into LDS ----
  {
    const __bf16* qrow = qg + ((size_t)(b * 2048 + s0)) * 1024;
    const int r = tid >> 4, cc = tid & 15;
    const v8bf* src = (const v8bf*)(qrow + r * 1024 + cc * 64);
    v8bf* dst = (v8bf*)(&sm.qtile[r][cc * 64]);
#pragma unroll
    for (int i = 0; i < 8; ++i) dst[i] = src[i];
  }
  __syncthreads();

  // ---- phase 1: scores; wave w covers key cols [w*256, w*256+256) ----
  const __bf16* kb_ = kg + (size_t)b * 2048 * 1024;
  auto ldQ = [&](int d) -> v16bf {
    return ld_frag(&sm.qtile[m_ln][d + kb16], &sm.qtile[m_ln][d + kb16 + 8]);
  };
  for (int jt = 0; jt < 16; jt += 2) {
    const int j0 = (w * 16 + jt) << 4;
    const __bf16* krow0 = kb_ + (size_t)(j0 + m_ln) * 1024 + kb16;
    const __bf16* krow1 = krow0 + 16 * 1024;
    v8f acc0 = vzero8(), acc1 = vzero8();
    v16bf a  = ldQ(0);
    v16bf b0 = ld_frag(krow0, krow0 + 8);
    v16bf b1 = ld_frag(krow1, krow1 + 8);
    for (int d0 = 0; d0 < 1024; d0 += 32) {
      const int dn = (d0 + 32) & 1023;
      v16bf a_n = ldQ(dn);
      v16bf b0n = ld_frag(krow0 + dn, krow0 + dn + 8);
      v16bf b1n = ld_frag(krow1 + dn, krow1 + dn + 8);
      acc0 = wmma_bf16(a, b0, acc0);
      acc1 = wmma_bf16(a, b1, acc1);
      a = a_n; b0 = b0n; b1 = b1n;
    }
    const int ja = j0 + m_ln, jc = j0 + 16 + m_ln;
#pragma unroll
    for (int r = 0; r < 8; ++r) {
      sm.probs[rb + r][ja] = (__bf16)acc0[r];
      sm.probs[rb + r][jc] = (__bf16)acc1[r];
    }
  }
  __syncthreads();

  // ---- phase 2: softmax; 16 threads per query row, 128 cols each ----
  {
    const int r = tid & 15, c = tid >> 4;
    const int jb = c * 128;
    float mx = -3.0e38f;
#pragma unroll 8
    for (int jj = 0; jj < 128; ++jj)
      mx = fmaxf(mx, (float)sm.probs[r][jb + jj]);
    sm.red[r][c] = mx;
    __syncthreads();
    if (tid < 16) {
      float m2 = sm.red[tid][0];
      for (int c2 = 1; c2 < 16; ++c2) m2 = fmaxf(m2, sm.red[tid][c2]);
      sm.rowmax[tid] = m2;
    }
    __syncthreads();
    const float rm = sm.rowmax[r];
    float ss = 0.f;
#pragma unroll 8
    for (int jj = 0; jj < 128; ++jj) {
      const float e = __expf((float)sm.probs[r][jb + jj] - rm);
      ss += e;
      sm.probs[r][jb + jj] = (__bf16)e;  // unnormalized; 1/sum folded later
    }
    sm.red[r][c] = ss;
    __syncthreads();
    if (tid < 16) {
      float s = 0.f;
      for (int c2 = 0; c2 < 16; ++c2) s += sm.red[tid][c2];
      sm.rowinv[tid] = 1.0f / s;
    }
    __syncthreads();
  }

  // ---- phase 3: PV; wave w covers h cols [w*128, w*128+128) ----
  const int h0b = w << 7;
  const __bf16* vTb = vTg + (size_t)b * 1024 * 2048;
  auto ldP = [&](int j) -> v16bf {
    return ld_frag(&sm.probs[m_ln][j + kb16], &sm.probs[m_ln][j + kb16 + 8]);
  };
  auto ldV = [&](int ht, int j) -> v16bf {
    const __bf16* vrow = vTb + (size_t)(h0b + ht * 16 + m_ln) * 2048 + j + kb16;
    return ld_frag(vrow, vrow + 8);
  };
  v8f acc[8];
#pragma unroll
  for (int i = 0; i < 8; ++i) acc[i] = vzero8();

  v16bf a = ldP(0);
  v16bf bcur = ldV(0, 0);
  for (int j0 = 0; j0 < 2048; j0 += 32) {
    const int jn = (j0 + 32) & 2047;   // wrap keeps prefetch in-bounds
    v16bf a_n = ldP(jn);
#pragma unroll
    for (int ht = 0; ht < 8; ++ht) {
      v16bf b_n = (ht < 7) ? ldV(ht + 1, j0) : ldV(0, jn);
      acc[ht] = wmma_bf16(a, bcur, acc[ht]);
      bcur = b_n;
    }
    a = a_n;
  }

  float inv[8];
#pragma unroll
  for (int r = 0; r < 8; ++r) inv[r] = sm.rowinv[rb + r];
  float* op = outg + ((size_t)(b * 2048 + s0)) * 1024;
#pragma unroll
  for (int ht = 0; ht < 8; ++ht) {
    const int h = h0b + ht * 16 + m_ln;  // lanes 0-15 / 16-31 -> coalesced 64B
#pragma unroll
    for (int r = 0; r < 8; ++r)
      op[(size_t)(rb + r) * 1024 + h] = acc[ht][r] * inv[r];
  }
}

// ---------------------------------------------------------------------------
// Host launcher
// ---------------------------------------------------------------------------
extern "C" void kernel_launch(void* const* d_in, const int* in_sizes, int n_in,
                              void* d_out, int out_size, void* d_ws,
                              size_t ws_size, hipStream_t stream) {
  (void)in_sizes; (void)n_in; (void)out_size; (void)ws_size;
  const float* x  = (const float*)d_in[0];
  const float* Wq = (const float*)d_in[1];
  const float* bq = (const float*)d_in[2];
  const float* Wk = (const float*)d_in[3];
  const float* bk = (const float*)d_in[4];
  const float* Wv = (const float*)d_in[5];
  const float* bv = (const float*)d_in[6];
  float* out = (float*)d_out;

  char* ws = (char*)d_ws;
  const size_t SZ_X = (size_t)16384 * 1024 * 2;  // 32 MiB per bf16 [16384,1024]
  const size_t SZ_W = (size_t)1024 * 1024 * 2;   // 2 MiB per bf16 [1024,1024]
  __bf16* xb  = (__bf16*)(ws);
  __bf16* wtq = (__bf16*)(ws + SZ_X);
  __bf16* wtk = (__bf16*)(ws + SZ_X + 1 * SZ_W);
  __bf16* wtv = (__bf16*)(ws + SZ_X + 2 * SZ_W);
  __bf16* qb  = (__bf16*)(ws + 1 * SZ_X + 3 * SZ_W);
  __bf16* kbf = (__bf16*)(ws + 2 * SZ_X + 3 * SZ_W);
  __bf16* vT  = (__bf16*)(ws + 3 * SZ_X + 3 * SZ_W);
  // total workspace used: 4*32MiB + 3*2MiB = 134 MiB

  conv_x_kernel<<<16384, 256, 0, stream>>>(x, xb);
  dim3 tb(32, 8), tg(32, 32);
  conv_wt_kernel<<<tg, tb, 0, stream>>>(Wq, wtq);
  conv_wt_kernel<<<tg, tb, 0, stream>>>(Wk, wtk);
  conv_wt_kernel<<<tg, tb, 0, stream>>>(Wv, wtv);
  qk_proj_kernel<<<2048, 256, 0, stream>>>(xb, wtq, bq, qb, 0.03125f);  // q * 1/sqrt(1024)
  qk_proj_kernel<<<2048, 256, 0, stream>>>(xb, wtk, bk, kbf, 1.0f);
  v_proj_kernel<<<2048, 256, 0, stream>>>(xb, wtv, bv, vT);
  attn_fused_kernel<<<1024, 256, 0, stream>>>(qb, kbf, vT, out);
}
